// LSTMClassifier_70592082477252
// MI455X (gfx1250) — matrix operand order repaired and checked by reading, not compiled, not run on hardware
//
#include <hip/hip_runtime.h>
#include <hip/hip_bf16.h>

#define HDIM 64
#define TLEN 512
#define NCLS 3
#define BTILE 16
#define XCHUNK 64

typedef __attribute__((ext_vector_type(16))) __bf16 v16bf;
typedef __attribute__((ext_vector_type(8)))  __bf16 v8bf;
typedef __attribute__((ext_vector_type(8)))  float  v8f;
typedef __attribute__((ext_vector_type(4)))  float  v4f;

union BU { uint4 u; v8bf b; };

__device__ __forceinline__ v16bf cat8(v8bf lo, v8bf hi) {
  return __builtin_shufflevector(lo, hi, 0,1,2,3,4,5,6,7,8,9,10,11,12,13,14,15);
}

__device__ __forceinline__ v8f wmma_bf16(v16bf a, v16bf b, v8f c) {
  return __builtin_amdgcn_wmma_f32_16x16x32_bf16(
      /*neg_a=*/false, a, /*neg_b=*/false, b,
      /*c_mod=*/(short)0, c, /*reuse_a=*/false, /*reuse_b=*/false);
}

__device__ __forceinline__ float sigm(float x) {
  return __builtin_amdgcn_rcpf(1.0f + __expf(-x));
}
__device__ __forceinline__ float tanh_(float x) {
  x = fminf(fmaxf(x, -15.0f), 15.0f);
  float e = __expf(2.0f * x);
  return (e - 1.0f) * __builtin_amdgcn_rcpf(e + 1.0f);
}

__global__ __launch_bounds__(128) void lstm2_fused_kernel(
    const float* __restrict__ x,
    const float* __restrict__ w_ih0, const float* __restrict__ w_hh0,
    const float* __restrict__ b_ih0, const float* __restrict__ b_hh0,
    const float* __restrict__ w_ih1, const float* __restrict__ w_hh1,
    const float* __restrict__ b_ih1, const float* __restrict__ b_hh1,
    const float* __restrict__ w_fc,  const float* __restrict__ b_fc,
    float* __restrict__ out)
{
  __shared__ __bf16 h0t[BTILE * HDIM];   // layer-0 hidden state tile (bf16, A-source)
  __shared__ __bf16 h1t[BTILE * HDIM];   // layer-1 hidden state tile
  __shared__ float  xs[XCHUNK * BTILE];  // x chunk, layout [tt][m]
  __shared__ float  h1f[BTILE * HDIM];   // final h1 in f32 for FC

  const int tid  = threadIdx.x;
  const int wv   = tid >> 5;      // wave id 0..3 -> hidden tile column group
  const int lane = tid & 31;
  const int n    = lane & 15;     // N / M index within tile
  const int half = lane >> 4;     // lane half select
  const int b0   = blockIdx.x * BTILE;

  // zero hidden tiles
  for (int i = tid; i < BTILE * HDIM; i += 128) {
    h0t[i] = (__bf16)0.0f;
    h1t[i] = (__bf16)0.0f;
  }

  // per-lane gate-column constants (4 gate groups: i,f,g,o)
  int col[4];
  float w0s[4], b0s[4], b1s[4];
#pragma unroll
  for (int g = 0; g < 4; ++g) {
    col[g] = g * 64 + wv * 16 + n;
    w0s[g] = w_ih0[col[g]];                       // input dim is 1
    b0s[g] = b_ih0[col[g]] + b_hh0[col[g]];
    b1s[g] = b_ih1[col[g]] + b_hh1[col[g]];
  }

  // B-matrix fragments (weights), resident in registers for the whole loop.
  // B[k][nn] = W[colbase+nn][k]; lane nn = lane&15; lanes 0-15: K=kb*32+0..15,
  // lanes 16-31: K=kb*32+16..31, packed 2 bf16 per VGPR.
  v16bf Bh0[4][2], Bi1[4][2], Bh1[4][2];
#pragma unroll
  for (int g = 0; g < 4; ++g) {
#pragma unroll
    for (int kb = 0; kb < 2; ++kb) {
      const int off = col[g] * HDIM + kb * 32 + half * 16;
      const float* p0 = w_hh0 + off;
      const float* p1 = w_ih1 + off;
      const float* p2 = w_hh1 + off;
      v16bf t0, t1, t2;
#pragma unroll
      for (int i = 0; i < 16; ++i) {
        t0[i] = (__bf16)p0[i];
        t1[i] = (__bf16)p1[i];
        t2[i] = (__bf16)p2[i];
      }
      Bh0[g][kb] = t0; Bi1[g][kb] = t1; Bh1[g][kb] = t2;
    }
  }

  // cell states (fragment layout: component r <-> batch row r + 8*half)
  v8f c0, c1;
#pragma unroll
  for (int r = 0; r < 8; ++r) { c0[r] = 0.0f; c1[r] = 0.0f; }
  float h1v[8];
#pragma unroll
  for (int r = 0; r < 8; ++r) h1v[r] = 0.0f;

  // A-fragment loader from an LDS h-tile (16-bit A layout, 16x32 per kblock)
  auto lda = [&](const __bf16* ht, int kb) -> v16bf {
    const __bf16* p = ht + (size_t)n * HDIM + kb * 32 + half * 16;
    BU lo, hi;
    lo.u = *(const uint4*)(p);
    hi.u = *(const uint4*)(p + 8);
    return cat8(lo.b, hi.b);
  };

  __syncthreads();

  for (int t = 0; t < TLEN; ++t) {
    if ((t & (XCHUNK - 1)) == 0) {
      // refill x chunk, transposed: xs[tt*16+m] = x[b0+m][t+tt]  (coalesced in tt)
      for (int i = tid; i < XCHUNK * BTILE; i += 128) {
        int m = i >> 6, tt = i & 63;
        xs[tt * BTILE + m] = x[(size_t)(b0 + m) * TLEN + t + tt];
      }
      __syncthreads();
    }

    // read previous h0 as A fragments
    v16bf a0[2];
    a0[0] = lda(h0t, 0);
    a0[1] = lda(h0t, 1);

    // x values for this step (batch rows half*8 .. half*8+7)
    const float* xp = xs + (t & (XCHUNK - 1)) * BTILE + half * 8;
    v4f xlo = *(const v4f*)(xp);
    v4f xhi = *(const v4f*)(xp + 4);
    float xv[8] = { xlo[0], xlo[1], xlo[2], xlo[3], xhi[0], xhi[1], xhi[2], xhi[3] };

    // ---- layer 0: gates = x*w_ih0 + bias + h0 @ w_hh0^T
    v8f acc[4];
#pragma unroll
    for (int g = 0; g < 4; ++g) {
      v8f a;
#pragma unroll
      for (int r = 0; r < 8; ++r) a[r] = xv[r] * w0s[g] + b0s[g];
      a = wmma_bf16(a0[0], Bh0[g][0], a);
      a = wmma_bf16(a0[1], Bh0[g][1], a);
      acc[g] = a;
    }
    float h0v[8];
#pragma unroll
    for (int r = 0; r < 8; ++r) {
      float iv = sigm(acc[0][r]);
      float fv = sigm(acc[1][r]);
      float gv = tanh_(acc[2][r]);
      float ov = sigm(acc[3][r]);
      float cv = fv * c0[r] + iv * gv;
      c0[r] = cv;
      h0v[r] = ov * tanh_(cv);
    }

    __syncthreads();  // all reads of h0t done

    // publish new h0 (bf16) to LDS
    {
      __bf16* dst = h0t + (size_t)(half * 8) * HDIM + wv * 16 + n;
#pragma unroll
      for (int r = 0; r < 8; ++r) dst[(size_t)r * HDIM] = (__bf16)h0v[r];
    }
    __syncthreads();  // new h0 visible

    // ---- layer 1: gates = bias + h0_new @ w_ih1^T + h1 @ w_hh1^T
    v16bf a0n[2], a1[2];
    a0n[0] = lda(h0t, 0);
    a0n[1] = lda(h0t, 1);
    a1[0]  = lda(h1t, 0);
    a1[1]  = lda(h1t, 1);

#pragma unroll
    for (int g = 0; g < 4; ++g) {
      v8f a;
#pragma unroll
      for (int r = 0; r < 8; ++r) a[r] = b1s[g];
      a = wmma_bf16(a0n[0], Bi1[g][0], a);
      a = wmma_bf16(a0n[1], Bi1[g][1], a);
      a = wmma_bf16(a1[0],  Bh1[g][0], a);
      a = wmma_bf16(a1[1],  Bh1[g][1], a);
      acc[g] = a;
    }
#pragma unroll
    for (int r = 0; r < 8; ++r) {
      float iv = sigm(acc[0][r]);
      float fv = sigm(acc[1][r]);
      float gv = tanh_(acc[2][r]);
      float ov = sigm(acc[3][r]);
      float cv = fv * c1[r] + iv * gv;
      c1[r] = cv;
      h1v[r] = ov * tanh_(cv);
    }
    // publish new h1 (reads of h1t this step happened above)
    {
      __bf16* dst = h1t + (size_t)(half * 8) * HDIM + wv * 16 + n;
#pragma unroll
      for (int r = 0; r < 8; ++r) dst[(size_t)r * HDIM] = (__bf16)h1v[r];
    }
    __syncthreads();  // h0t/h1t/xs safe for next iteration
  }

  // ---- FC head on final h1
  {
    float* dst = h1f + (size_t)(half * 8) * HDIM + wv * 16 + n;
#pragma unroll
    for (int r = 0; r < 8; ++r) dst[(size_t)r * HDIM] = h1v[r];
  }
  __syncthreads();
  if (tid < BTILE * NCLS) {
    int m = tid / NCLS, c = tid % NCLS;
    float s = b_fc[c];
    const float* hr = h1f + (size_t)m * HDIM;
    const float* wr = w_fc + (size_t)c * HDIM;
#pragma unroll 8
    for (int k = 0; k < HDIM; ++k) s += hr[k] * wr[k];
    out[(size_t)(b0 + m) * NCLS + c] = s;
  }
}

extern "C" void kernel_launch(void* const* d_in, const int* in_sizes, int n_in,
                              void* d_out, int out_size, void* d_ws, size_t ws_size,
                              hipStream_t stream) {
  const float* x     = (const float*)d_in[0];
  const float* w_ih0 = (const float*)d_in[1];
  const float* w_hh0 = (const float*)d_in[2];
  const float* b_ih0 = (const float*)d_in[3];
  const float* b_hh0 = (const float*)d_in[4];
  const float* w_ih1 = (const float*)d_in[5];
  const float* w_hh1 = (const float*)d_in[6];
  const float* b_ih1 = (const float*)d_in[7];
  const float* b_hh1 = (const float*)d_in[8];
  const float* w_fc  = (const float*)d_in[9];
  const float* b_fc  = (const float*)d_in[10];
  float* out = (float*)d_out;

  const int B = 2048;
  dim3 grid(B / BTILE), block(128);
  lstm2_fused_kernel<<<grid, block, 0, stream>>>(
      x, w_ih0, w_hh0, b_ih0, b_hh0,
      w_ih1, w_hh1, b_ih1, b_hh1, w_fc, b_fc, out);
}
